// _NoisyPolygonGraphTrunk_23802708754998
// MI455X (gfx1250) — matrix operand order repaired and checked by reading, not compiled, not run on hardware
//
#include <hip/hip_runtime.h>

// ---------------------------------------------------------------------------
// NoisyPolygonGraphTrunk — fused CDNA5 (gfx1250) kernel.
//
// Shapes: B=1024 graphs, n=256 nodes, HID=256, TDIM=128, NODE_IN=134.
// ~87 GFLOP total, ~4 MB DRAM I/O -> compute bound -> v_wmma_f32_16x16x32_f16.
//
// Linearity trick: (s + roll(s,1) + roll(s,-1))/3 with s = h@W equals
// ((h + roll(h,1) + roll(h,-1))/3) @ W, so the ring average runs on h
// (packed-f16 VALU in LDS) and bias+SiLU fuse into the WMMA epilogue.
//
// One workgroup (8 wave32) per graph. h lives in LDS (f16, row stride 264 to
// avoid 16-way bank conflicts on A-fragment loads), ping-ponging between two
// buffers. Weights are pre-packed into a contiguous-K WMMA B-fragment layout so every
// 32x16 B tile is one contiguous 1KB block (two global_load_b128 per lane),
// L2-resident across all 1024 blocks. Each wave runs TWO independent WMMA
// accumulation chains (dual N-tiles) to keep the matrix pipe busy at the
// 2-waves/SIMD occupancy this LDS footprint implies.
// ---------------------------------------------------------------------------

typedef __attribute__((ext_vector_type(16))) _Float16 v16h;
typedef __attribute__((ext_vector_type(8)))  _Float16 h8;
typedef __attribute__((ext_vector_type(8)))  float    v8f;

#define NB      1024
#define NNODE   256
#define HID     256
#define TDIM    128
#define STR     264          // padded f16 row stride (132 dwords -> bank shift 4/row)
#define KT0     5            // layer-0 K tiles: NODE_IN=134 padded to 160
#define KT1     8            // layers 1/2: K=256
#define NT      16           // N tiles (256/16)
#define FRAG_ELEMS 512       // 32 lanes * 16 f16 per B fragment

// packed-weight offsets in d_ws (f16 units)
#define WP0_OFF 0
#define WP1_OFF (KT0 * NT * FRAG_ELEMS)              // 40960
#define WP2_OFF (WP1_OFF + KT1 * NT * FRAG_ELEMS)    // 106496

// LDS layout (bytes)
#define HBUF_BYTES (NNODE * STR * 2)                 // 135168
#define LDS_HA     0
#define LDS_HB     (HBUF_BYTES)
#define LDS_TEMB   (2 * HBUF_BYTES)                  // 128 f32
#define LDS_E      (LDS_TEMB + 512)                  // 128 f32
#define LDS_BIAS   (LDS_E + 512)                     // 256 f32
#define LDS_TOTAL  (LDS_BIAS + 1024)                 // 272384 <= 320KB/WG

// fast sigmoid: v_rcp_f32 instead of IEEE divide (we already run f16 WMMA)
__device__ __forceinline__ float fast_silu(float v) {
  return v * __builtin_amdgcn_rcpf(1.0f + __expf(-v));
}

// ---------------------------------------------------------------------------
// Pack W (f32, [K x 256] row-major) into a contiguous-K WMMA B-fragment layout (f16):
// fragment (kt, nt) is 512 contiguous f16; lane L holds the 16 contiguous
// K-values { kt*32 + (L<16 ? 0 : 16) + p } of column nt*16 + (L & 15).
// Rows k >= Ksrc are zero (layer-0 padding 134 -> 160).
// ---------------------------------------------------------------------------
__global__ void __launch_bounds__(256)
polytrunk_pack_w(const float* __restrict__ W0, const float* __restrict__ W1,
                 const float* __restrict__ W2, _Float16* __restrict__ wp) {
  const int layer = blockIdx.y;
  const int idx = blockIdx.x * 256 + threadIdx.x;

  const float* W;
  int Ksrc, KT; int base;
  if (layer == 0)      { W = W0; Ksrc = 134; KT = KT0; base = WP0_OFF; }
  else if (layer == 1) { W = W1; Ksrc = 256; KT = KT1; base = WP1_OFF; }
  else                 { W = W2; Ksrc = 256; KT = KT1; base = WP2_OFF; }

  if (idx >= KT * NT * FRAG_ELEMS) return;

  const int frag   = idx >> 9;          // 512 elems per fragment
  const int within = idx & 511;
  const int lane   = within >> 4;
  const int p      = within & 15;
  const int kt     = frag >> 4;
  const int nt     = frag & 15;

  const int k   = kt * 32 + ((lane >> 4) << 4) + p;   // lanes 16-31 -> K+16
  const int col = nt * 16 + (lane & 15);
  const float v = (k < Ksrc) ? W[k * HID + col] : 0.0f;
  wp[base + idx] = (_Float16)v;
}

// ---------------------------------------------------------------------------
// Ring average: dst[i] = (src[i-1] + src[i] + src[i+1]) / 3   (cyclic)
// thread == node; packed-f16 VALU (v_pk_add_f16 / v_pk_mul_f16).
// ---------------------------------------------------------------------------
__device__ __forceinline__ void ring_avg(const _Float16* __restrict__ src,
                                         _Float16* __restrict__ dst, int tid) {
  const int i  = tid;
  const int ip = (i + NNODE - 1) & (NNODE - 1);
  const int in = (i + 1) & (NNODE - 1);
  const h8* rp = (const h8*)(src + ip * STR);
  const h8* rc = (const h8*)(src + i  * STR);
  const h8* rn = (const h8*)(src + in * STR);
  h8*       rd = (h8*)(dst + i * STR);
  const _Float16 third = (_Float16)(1.0f / 3.0f);
  #pragma unroll 4
  for (int c = 0; c < HID / 8; ++c) {
    h8 s = rp[c] + rc[c] + rn[c];      // v_pk_add_f16 x8
    rd[c] = s * third;                 // v_pk_mul_f16 x4 (splat)
  }
}

// ---------------------------------------------------------------------------
// GEMM + bias + SiLU epilogue:  dst = silu(src @ Wpack + bias)
// src: [256 x K<=256] f16 in LDS (stride STR).  Each wave owns 2 M-tiles x
// 16 N-tiles, processed as 8 pairs of N-tiles with dual accumulators (two
// independent WMMA chains per wave).  A fragments use the contiguous-K 16-bit A
// layout: lane L (L<16: row m0+L, K = kt*32+{0..15}; L>=16: row
// m0+L-16, K = kt*32+{16..31}).
// ---------------------------------------------------------------------------
template <int KT>
__device__ __forceinline__ void gemm_silu(const _Float16* __restrict__ src,
                                          _Float16* __restrict__ dst,
                                          const _Float16* __restrict__ wp,
                                          const float* __restrict__ biasLds,
                                          int lane, int wave) {
  const int lm = lane & 15;
  const int lh = lane >> 4;                    // 0 or 1
  #pragma unroll
  for (int mi = 0; mi < 2; ++mi) {
    const int m0  = ((wave << 1) | mi) << 4;   // M-tile base node
    const int row = m0 + lm;

    v16h a[KT];
    #pragma unroll
    for (int kt = 0; kt < KT; ++kt) {
      const _Float16* p = src + row * STR + kt * 32 + lh * 16;
      h8 lo = *(const h8*)p;                   // K = kt*32 + lh*16 + [0..7]
      h8 hi = *(const h8*)(p + 8);             // K = kt*32 + lh*16 + 8 + [0..7]
      a[kt] = __builtin_shufflevector(lo, hi, 0, 1, 2, 3, 4, 5, 6, 7,
                                              8, 9, 10, 11, 12, 13, 14, 15);
    }

    for (int nt = 0; nt < NT; nt += 2) {
      v8f acc0 = {0.f, 0.f, 0.f, 0.f, 0.f, 0.f, 0.f, 0.f};
      v8f acc1 = {0.f, 0.f, 0.f, 0.f, 0.f, 0.f, 0.f, 0.f};
      #pragma unroll
      for (int kt = 0; kt < KT; ++kt) {
        const _Float16* f0 = wp + ((kt << 4) + nt) * FRAG_ELEMS + lane * 16;
        const _Float16* f1 = f0 + FRAG_ELEMS;
        h8 b0lo = *(const h8*)f0;
        h8 b0hi = *(const h8*)(f0 + 8);
        h8 b1lo = *(const h8*)f1;
        h8 b1hi = *(const h8*)(f1 + 8);
        v16h bm0 = __builtin_shufflevector(b0lo, b0hi, 0, 1, 2, 3, 4, 5, 6, 7,
                                                       8, 9, 10, 11, 12, 13, 14, 15);
        v16h bm1 = __builtin_shufflevector(b1lo, b1hi, 0, 1, 2, 3, 4, 5, 6, 7,
                                                       8, 9, 10, 11, 12, 13, 14, 15);
        acc0 = __builtin_amdgcn_wmma_f32_16x16x32_f16(
            false, a[kt], false, bm0, (short)0, acc0, false, false);
        acc1 = __builtin_amdgcn_wmma_f32_16x16x32_f16(
            false, a[kt], false, bm1, (short)0, acc1, false, false);
      }
      // C/D layout: VGPR r, lane L -> M = m0 + r + (L>=16)*8, N = nt*16 + L%16
      const int   col0  = (nt << 4) + lm;
      const int   col1  = col0 + 16;
      const float bias0 = biasLds[col0];
      const float bias1 = biasLds[col1];
      const int   rb    = m0 + lh * 8;
      #pragma unroll
      for (int r = 0; r < 8; ++r) {
        dst[(rb + r) * STR + col0] = (_Float16)fast_silu(acc0[r] + bias0);
        dst[(rb + r) * STR + col1] = (_Float16)fast_silu(acc1[r] + bias1);
      }
    }
  }
}

// ---------------------------------------------------------------------------
// Main fused kernel: one block per graph.
// ---------------------------------------------------------------------------
__global__ void __launch_bounds__(256)
polytrunk_fused(const float* __restrict__ x, const float* __restrict__ t,
                const float* __restrict__ time_w, const float* __restrict__ time_b,
                const float* __restrict__ b0, const float* __restrict__ b1,
                const float* __restrict__ b2, const _Float16* __restrict__ wp,
                float* __restrict__ out) {
  extern __shared__ __align__(16) char smem[];
  _Float16* hA      = (_Float16*)(smem + LDS_HA);
  _Float16* hB      = (_Float16*)(smem + LDS_HB);
  float*    tembLds = (float*)(smem + LDS_TEMB);
  float*    eLds    = (float*)(smem + LDS_E);
  float*    biasLds = (float*)(smem + LDS_BIAS);

  const int tid  = threadIdx.x;
  const int lane = tid & 31;
  const int wave = tid >> 5;
  const int b    = blockIdx.x;

  // ---- Phase 0: sinusoidal(t) ----------------------------------------------
  const float tb = t[b];
  if (tid < TDIM) {
    const int half = TDIM / 2;                        // 64
    const int i    = (tid < half) ? tid : tid - half;
    const float fr = __expf((float)i * (-9.210340372f / (float)(half - 1)));
    eLds[tid] = (tid < half) ? __sinf(tb * fr) : __cosf(tb * fr);
  }
  __syncthreads();

  // ---- time MLP (silu(e @ time_w + time_b)) + coords/PE/zero-pad -----------
  if (tid < TDIM) {
    float s = time_b[tid];
    #pragma unroll 8
    for (int k = 0; k < TDIM; ++k) s += eLds[k] * time_w[k * TDIM + tid];
    tembLds[tid] = fast_silu(s);
  }
  {
    const int i = tid;                                // node
    _Float16* hr = hA + i * STR;
    const float x0 = x[b * (NNODE * 2) + 2 * i];
    const float x1 = x[b * (NNODE * 2) + 2 * i + 1];
    const float th = 6.2831853071795864f * (float)i / (float)NNODE;
    hr[0] = (_Float16)x0;
    hr[1] = (_Float16)x1;
    hr[2] = (_Float16)__sinf(th);
    hr[3] = (_Float16)__cosf(th);
    hr[4] = (_Float16)__sinf(2.0f * th);
    hr[5] = (_Float16)__cosf(2.0f * th);
    // zero K-pad: cols 134..255 (h8-vectorized from 136, 16B-aligned)
    hr[134] = (_Float16)0.0f;
    hr[135] = (_Float16)0.0f;
    const h8 z = {};
    #pragma unroll
    for (int c = 136; c < HID; c += 8) *(h8*)(hr + c) = z;
  }
  biasLds[tid] = b0[tid];
  __syncthreads();
  {   // broadcast time embedding into every node row
    _Float16* hr = hA + tid * STR + 6;
    #pragma unroll 8
    for (int j = 0; j < TDIM; ++j) hr[j] = (_Float16)tembLds[j];
  }
  __syncthreads();

  // ---- Layer 0 : avg(hA)->hB ; silu(hB@W0+b0)->hA --------------------------
  ring_avg(hA, hB, tid);
  __syncthreads();
  gemm_silu<KT0>(hB, hA, wp + WP0_OFF, biasLds, lane, wave);
  __syncthreads();

  // ---- Layer 1 -------------------------------------------------------------
  ring_avg(hA, hB, tid);
  biasLds[tid] = b1[tid];
  __syncthreads();
  gemm_silu<KT1>(hB, hA, wp + WP1_OFF, biasLds, lane, wave);
  __syncthreads();

  // ---- Layer 2 -------------------------------------------------------------
  ring_avg(hA, hB, tid);
  biasLds[tid] = b2[tid];
  __syncthreads();
  gemm_silu<KT1>(hB, hA, wp + WP2_OFF, biasLds, lane, wave);
  __syncthreads();

  // ---- Mean pool over nodes ------------------------------------------------
  {
    float s = 0.0f;
    #pragma unroll 8
    for (int i = 0; i < NNODE; ++i) s += (float)hA[i * STR + tid];
    out[b * HID + tid] = s * (1.0f / (float)NNODE);
  }
}

// ---------------------------------------------------------------------------
// Host launcher.
// Inputs: x, t, time_w, time_b, W0, b0, W1, b1, W2, b2 (all f32).
// ---------------------------------------------------------------------------
extern "C" void kernel_launch(void* const* d_in, const int* in_sizes, int n_in,
                              void* d_out, int out_size, void* d_ws, size_t ws_size,
                              hipStream_t stream) {
  const float* x      = (const float*)d_in[0];
  const float* t      = (const float*)d_in[1];
  const float* time_w = (const float*)d_in[2];
  const float* time_b = (const float*)d_in[3];
  const float* W0     = (const float*)d_in[4];
  const float* b0     = (const float*)d_in[5];
  const float* W1     = (const float*)d_in[6];
  const float* b1     = (const float*)d_in[7];
  const float* W2     = (const float*)d_in[8];
  const float* b2     = (const float*)d_in[9];
  float*       out    = (float*)d_out;
  _Float16*    wp     = (_Float16*)d_ws;   // 344064 B of packed f16 weights

  // Pack weights into WMMA B-fragment layout (every call; deterministic).
  dim3 pgrid((KT1 * NT * FRAG_ELEMS + 255) / 256, 3);
  polytrunk_pack_w<<<pgrid, 256, 0, stream>>>(W0, W1, W2, wp);

  polytrunk_fused<<<NB, 256, LDS_TOTAL, stream>>>(x, t, time_w, time_b,
                                                  b0, b1, b2, wp, out);
}
